// AttentionCl_63453846831631
// MI455X (gfx1250) — compile-verified, matrix operand order checked by reading, not run on hardware
//
#include <hip/hip_runtime.h>
#include <hip/hip_bf16.h>

typedef __attribute__((ext_vector_type(4)))  _Float16 v4h;
typedef __attribute__((ext_vector_type(8)))  _Float16 v8h;
typedef __attribute__((ext_vector_type(16))) _Float16 v16h;
typedef __attribute__((ext_vector_type(8)))  float    v8f;

#define NB 8
#define NSEQ 1024
#define CDIM 768
#define DH 32
#define NH 24            // CDIM / DH
#define ROWS (NB * NSEQ) // 8192

// Build a 16-bit WMMA A/B operand for one lane.
// Layout (CDNA5 ISA 7.12.2): lanes 0-15 hold K {0..7, 16..23}, lanes 16-31 hold
// K {8..15, 24..31}. Caller passes p already offset by (lane>>4)*8 elements;
// the two contiguous 8-element runs are at p and p+16.
__device__ __forceinline__ v16h make_ab(const _Float16* p) {
    v8h lo = *(const v8h*)(p);
    v8h hi = *(const v8h*)(p + 16);
    v16h r;
#pragma unroll
    for (int i = 0; i < 8; ++i) { r[i] = lo[i]; r[i + 8] = hi[i]; }
    return r;
}

__device__ __forceinline__ v8f wmma_f16(v16h a, v16h b, v8f c) {
    return __builtin_amdgcn_wmma_f32_16x16x32_f16(false, a, false, b,
                                                  (short)0, c, false, false);
}

// ---------------- fp32 -> fp16 conversion (vector of 4) ----------------
__global__ void cvt_f32_f16_k(const float* __restrict__ s, _Float16* __restrict__ d, int n4) {
    int i = blockIdx.x * blockDim.x + threadIdx.x;
    if (i < n4) {
        float4 v = ((const float4*)s)[i];
        v4h h = { (_Float16)v.x, (_Float16)v.y, (_Float16)v.z, (_Float16)v.w };
        ((v4h*)d)[i] = h;
    }
}

// ---------------- QKV projection GEMM (32x64 tile per wave) ----------------
// y[m][o] = sum_c x[m][c] * Wqkv[o][c] + bqkv[o], scattered into head-first
// Q (pre-scaled by dh^-0.5), K row-major [bh][n][32], V transposed [bh][32][n].
// Operands are L2-resident; the loop is software-pipelined, no prefetch needed.
__global__ void qkv_gemm_k(const _Float16* __restrict__ xh,
                           const _Float16* __restrict__ wh,
                           const float* __restrict__ bias,
                           _Float16* __restrict__ qh,
                           _Float16* __restrict__ kh,
                           _Float16* __restrict__ vt) {
    const int lane = threadIdx.x & 31;
    const int wv   = threadIdx.x >> 5;
    const int tile = blockIdx.x * 4 + wv;            // 256 * 36 = 9216 tiles
    const int mt = tile / 36, ot = tile % 36;
    const int m0 = mt * 32, o0 = ot * 64;
    const int r16 = lane & 15, hf = (lane >> 4) & 1;

    const _Float16* ap0 = xh + (size_t)(m0 + r16) * CDIM + hf * 8;
    const _Float16* ap1 = ap0 + (size_t)16 * CDIM;
    const _Float16* bp  = wh + (size_t)(o0 + r16) * CDIM + hf * 8;

    v8f acc[2][4] = {};
#pragma unroll 2
    for (int k = 0; k < CDIM; k += 32) {
        v16h a0 = make_ab(ap0 + k);
        v16h a1 = make_ab(ap1 + k);
#pragma unroll
        for (int j = 0; j < 4; ++j) {
            v16h b = make_ab(bp + (size_t)j * 16 * CDIM + k);
            acc[0][j] = wmma_f16(a0, b, acc[0][j]);
            acc[1][j] = wmma_f16(a1, b, acc[1][j]);
        }
    }

    const float scale = 0.1767766952966369f; // 32^-0.5
#pragma unroll
    for (int j = 0; j < 4; ++j) {
        const int o = o0 + j * 16 + r16;     // output channel for this lane
        const float bo = bias[o];
        const int h = o / 96, t = o % 96;    // head-first: [h][3*dh]
#pragma unroll
        for (int i = 0; i < 2; ++i) {
#pragma unroll
            for (int r = 0; r < 8; ++r) {
                int m = m0 + i * 16 + r + 8 * hf;
                int b_ = m >> 10, n = m & (NSEQ - 1);
                float v = acc[i][j][r] + bo;
                size_t bh = (size_t)(b_ * NH + h);
                if (t < 32) {
                    qh[(bh * NSEQ + n) * DH + t] = (_Float16)(v * scale);
                } else if (t < 64) {
                    kh[(bh * NSEQ + n) * DH + (t - 32)] = (_Float16)v;
                } else {
                    vt[(bh * DH + (t - 64)) * NSEQ + n] = (_Float16)v;
                }
            }
        }
    }
}

// ---------------- flash attention ----------------
// One wave per (b,h, 16-query tile). S via wmma (K=dh=32 in one shot),
// online softmax, P transposed through per-wave LDS tile, O via wmma.
// rel_pos is the only first-touch HBM stream (100.7 MB): stage it into GL2
// with one builtin prefetch per iteration (16 rows x 1 cacheline each;
// lanes 0-15 cover tile k0+32, lanes 16-31 cover tile k0+64).
__global__ void attn_k(const _Float16* __restrict__ qh,
                       const _Float16* __restrict__ kh,
                       const _Float16* __restrict__ vt,
                       const float* __restrict__ rel,
                       _Float16* __restrict__ oh) {
    __shared__ _Float16 lds[4 * 16 * 32];
    const int lane = threadIdx.x & 31;
    const int wv   = threadIdx.x >> 5;
    const int tile = blockIdx.x * 4 + wv;            // 12288 tiles
    const int qt = tile & 63;
    const int bh = tile >> 6;                        // b*NH + h
    const int h = bh % NH, b = bh / NH;
    const int q0 = qt * 16;
    const int r16 = lane & 15, hf = (lane >> 4) & 1;
    _Float16* plds = lds + wv * 512;

    // Q tile as A operand (resident for the whole loop)
    v16h Qa = make_ab(qh + ((size_t)bh * NSEQ + q0 + r16) * DH + hf * 8);

    v8f O0 = {}, O1 = {};
    float rmax[8], rsum[8];
#pragma unroll
    for (int r = 0; r < 8; ++r) { rmax[r] = -1e30f; rsum[r] = 0.0f; }

    const float*    relp  = rel + ((size_t)h * NSEQ + q0) * NSEQ;
    const _Float16* kbase = kh + (size_t)bh * NSEQ * DH;
    const _Float16* vbase = vt + (size_t)bh * DH * NSEQ;

    // per-lane prefetch cursor into the rel_pos bias stream
    const float* relpf = relp + (size_t)r16 * NSEQ + hf * 32;

    for (int k0 = 0; k0 < NSEQ; k0 += 32) {
        v16h Kb0 = make_ab(kbase + (size_t)(k0 + r16) * DH + hf * 8);
        v16h Kb1 = make_ab(kbase + (size_t)(k0 + 16 + r16) * DH + hf * 8);
        __builtin_prefetch(relpf + k0 + 32, 0, 3);   // HBM -> GL2 staging
        v8f z = {};
        v8f S0 = wmma_f16(Qa, Kb0, z);
        v8f S1 = wmma_f16(Qa, Kb1, z);
#pragma unroll
        for (int r = 0; r < 8; ++r) {
            int qrow = r + 8 * hf;                   // local row this lane holds
            float s0 = S0[r] + relp[(size_t)qrow * NSEQ + k0 + r16];
            float s1 = S1[r] + relp[(size_t)qrow * NSEQ + k0 + 16 + r16];
            float t = fmaxf(s0, s1);
#pragma unroll
            for (int m = 1; m < 16; m <<= 1) t = fmaxf(t, __shfl_xor(t, m));
            float nm = fmaxf(rmax[r], t);
            float corr = __expf(rmax[r] - nm);
            rmax[r] = nm;
            float p0 = __expf(s0 - nm);
            float p1 = __expf(s1 - nm);
            float ps = p0 + p1;
#pragma unroll
            for (int m = 1; m < 16; m <<= 1) ps += __shfl_xor(ps, m);
            rsum[r] = rsum[r] * corr + ps;
            O0[r] *= corr;
            O1[r] *= corr;
            plds[qrow * 32 + r16]      = (_Float16)p0;
            plds[qrow * 32 + 16 + r16] = (_Float16)p1;
        }
        __syncthreads();   // exact grid: every wave runs every iteration
        v16h Pa  = make_ab(plds + r16 * 32 + hf * 8);
        v16h Vb0 = make_ab(vbase + (size_t)r16 * NSEQ        + k0 + hf * 8);
        v16h Vb1 = make_ab(vbase + (size_t)(16 + r16) * NSEQ + k0 + hf * 8);
        O0 = wmma_f16(Pa, Vb0, O0);
        O1 = wmma_f16(Pa, Vb1, O1);
    }

#pragma unroll
    for (int r = 0; r < 8; ++r) {
        int qrow = q0 + r + 8 * hf;
        float inv = 1.0f / rsum[r];
        size_t base = ((size_t)b * NSEQ + qrow) * CDIM + h * DH;
        oh[base + r16]      = (_Float16)(O0[r] * inv);
        oh[base + 16 + r16] = (_Float16)(O1[r] * inv);
    }
}

// ---------------- output projection GEMM (32x64 tile per wave) ----------------
__global__ void proj_gemm_k(const _Float16* __restrict__ ah,
                            const _Float16* __restrict__ wh,
                            const float* __restrict__ bias,
                            float* __restrict__ out) {
    const int lane = threadIdx.x & 31;
    const int wv   = threadIdx.x >> 5;
    const int tile = blockIdx.x * 4 + wv;            // 256 * 12 = 3072 tiles
    const int mt = tile / 12, ot = tile % 12;
    const int m0 = mt * 32, o0 = ot * 64;
    const int r16 = lane & 15, hf = (lane >> 4) & 1;

    const _Float16* ap0 = ah + (size_t)(m0 + r16) * CDIM + hf * 8;
    const _Float16* ap1 = ap0 + (size_t)16 * CDIM;
    const _Float16* bp  = wh + (size_t)(o0 + r16) * CDIM + hf * 8;

    v8f acc[2][4] = {};
#pragma unroll 2
    for (int k = 0; k < CDIM; k += 32) {
        v16h a0 = make_ab(ap0 + k);
        v16h a1 = make_ab(ap1 + k);
#pragma unroll
        for (int j = 0; j < 4; ++j) {
            v16h b = make_ab(bp + (size_t)j * 16 * CDIM + k);
            acc[0][j] = wmma_f16(a0, b, acc[0][j]);
            acc[1][j] = wmma_f16(a1, b, acc[1][j]);
        }
    }

#pragma unroll
    for (int j = 0; j < 4; ++j) {
        const int o = o0 + j * 16 + r16;
        const float bo = bias[o];
#pragma unroll
        for (int i = 0; i < 2; ++i) {
#pragma unroll
            for (int r = 0; r < 8; ++r) {
                int m = m0 + i * 16 + r + 8 * hf;
                out[(size_t)m * CDIM + o] = acc[i][j][r] + bo;
            }
        }
    }
}

extern "C" void kernel_launch(void* const* d_in, const int* in_sizes, int n_in,
                              void* d_out, int out_size, void* d_ws, size_t ws_size,
                              hipStream_t stream) {
    const float* x     = (const float*)d_in[0];
    const float* rel   = (const float*)d_in[1];
    const float* Wqkv  = (const float*)d_in[2];
    const float* bqkv  = (const float*)d_in[3];
    const float* Wproj = (const float*)d_in[4];
    const float* bproj = (const float*)d_in[5];
    float* out = (float*)d_out;

    // workspace carve (256B aligned)
    char* w = (char*)d_ws;
    auto carve = [&](size_t bytes) -> void* {
        void* p = (void*)w;
        w += (bytes + 255) & ~(size_t)255;
        return p;
    };
    _Float16* xh  = (_Float16*)carve((size_t)ROWS * CDIM * 2);       // 12.6 MB
    _Float16* wqh = (_Float16*)carve((size_t)3 * CDIM * CDIM * 2);   //  3.5 MB
    _Float16* wph = (_Float16*)carve((size_t)CDIM * CDIM * 2);       //  1.2 MB
    _Float16* qh  = (_Float16*)carve((size_t)ROWS * CDIM * 2);       // 12.6 MB
    _Float16* kh  = (_Float16*)carve((size_t)ROWS * CDIM * 2);       // 12.6 MB
    _Float16* vt  = (_Float16*)carve((size_t)ROWS * CDIM * 2);       // 12.6 MB
    _Float16* oh  = (_Float16*)carve((size_t)ROWS * CDIM * 2);       // 12.6 MB
    (void)ws_size; (void)n_in; (void)in_sizes; (void)out_size;

    // 1) fp32 -> fp16 conversions
    {
        int n4 = ROWS * CDIM / 4;
        cvt_f32_f16_k<<<(n4 + 255) / 256, 256, 0, stream>>>(x, xh, n4);
        n4 = 3 * CDIM * CDIM / 4;
        cvt_f32_f16_k<<<(n4 + 255) / 256, 256, 0, stream>>>(Wqkv, wqh, n4);
        n4 = CDIM * CDIM / 4;
        cvt_f32_f16_k<<<(n4 + 255) / 256, 256, 0, stream>>>(Wproj, wph, n4);
    }

    // 2) QKV projection: (8192/32)*(2304/64) = 9216 wave tiles, 4 waves/block
    qkv_gemm_k<<<9216 / 4, 128, 0, stream>>>(xh, wqh, bqkv, qh, kh, vt);

    // 3) attention: 8*24*64 = 12288 wave tiles, 4 waves/block (exact grid)
    attn_k<<<12288 / 4, 128, 0, stream>>>(qh, kh, vt, rel, oh);

    // 4) output projection: (8192/32)*(768/64) = 3072 wave tiles
    proj_gemm_k<<<3072 / 4, 128, 0, stream>>>(oh, wph, bproj, out);
}
